// Inference_Sinkhorn_Loss_14800457302354
// MI455X (gfx1250) — compile-verified, matrix operand order checked by reading, not run on hardware
//
#include <hip/hip_runtime.h>

#define BS    16384
#define NC    1000
#define NCP   1008      // NC padded to multiple of 16 (zero rows) for branch-free GEMM
#define DIM   512
#define NITER 100

typedef __attribute__((ext_vector_type(16))) __bf16        v16bf;
typedef __attribute__((ext_vector_type(8)))  float         v8f;
typedef __attribute__((ext_vector_type(8)))  unsigned int  v8u;

// ---- fp32 -> bf16 (round-to-nearest-even), packed pair (used once, off hot path) ----
__device__ __forceinline__ unsigned int bf16bits(float f) {
    unsigned int u = __float_as_uint(f);
    u += 0x7FFFu + ((u >> 16) & 1u);
    return u >> 16;
}
__device__ __forceinline__ unsigned int bf16pk(float lo, float hi) {
    return bf16bits(lo) | (bf16bits(hi) << 16);
}

// Load a 16-bf16 WMMA fragment for this lane: 8 values at p+ka, 8 at p+kb.
// Pre-packed bf16 in memory -> two 16B loads, zero conversion VALU.
__device__ __forceinline__ v16bf load_frag(const unsigned short* __restrict__ p,
                                           int ka, int kb) {
    uint4 a = *reinterpret_cast<const uint4*>(p + ka);
    uint4 b = *reinterpret_cast<const uint4*>(p + kb);
    v8u r;
    r[0] = a.x; r[1] = a.y; r[2] = a.z; r[3] = a.w;
    r[4] = b.x; r[5] = b.y; r[6] = b.z; r[7] = b.w;
    return __builtin_bit_cast(v16bf, r);
}

// Gibbs kernel entry: dot -> sim = 2*dot ; M = sim/(1-sim) ; K = exp(-2*M)
__device__ __forceinline__ float gibbs(float dot) {
    float sim = dot * 2.0f;                 // / TEMPERATURE (0.5)
    float M   = sim / (1.0f - sim);
    return __expf(-2.0f * M);               // exp(M / -REG), REG = 0.5
}

// ---------------------------------------------------------------------------
// One-shot conversions (streamed once; VALU cost amortized)
// ---------------------------------------------------------------------------
__global__ __launch_bounds__(256)
void cvt_f_kernel(const float* __restrict__ F, unsigned int* __restrict__ Fb) {
    const int i = blockIdx.x * 256 + threadIdx.x;     // packed-pair index
    const float2 f = reinterpret_cast<const float2*>(F)[i];
    Fb[i] = bf16pk(f.x, f.y);
}

__global__ __launch_bounds__(256)
void cvt_t_kernel(const float* __restrict__ T, unsigned int* __restrict__ Tb) {
    const int i   = blockIdx.x * 256 + threadIdx.x;   // padded [NCP,256] pair index
    const int row = i >> 8;                           // 256 pairs per row (DIM/2)
    unsigned int o = 0u;                              // zero-pad rows >= NC
    if (row < NC) {
        const float2 f = reinterpret_cast<const float2*>(T)[i];
        o = bf16pk(f.x, f.y);
    }
    Tb[i] = o;
}

// ---------------------------------------------------------------------------
// GEMM + transform:  Km[i,j] = gibbs( F[i,:] . T[j,:] )  via WMMA bf16.
// block = 128 threads (4 waves); wave computes 64(M) x 16(N): 4 accumulators
// share one B fragment -> 4 WMMAs per 10 b128 loads per k-step.
// grid = (BS/256, NCP/16)
// ---------------------------------------------------------------------------
__global__ __launch_bounds__(128)
void gemm_gibbs_kernel(const unsigned short* __restrict__ Fb,
                       const unsigned short* __restrict__ Tb,
                       float* __restrict__ Km) {
    const int lane = threadIdx.x & 31;
    const int wave = threadIdx.x >> 5;
    const int m    = lane & 15;     // row within A tile / col within B tile
    const int kh   = lane >> 4;     // K-half select per WMMA lane layout

    const int rowBase = blockIdx.x * 256 + wave * 64;
    const int colBase = blockIdx.y * 16;

    const unsigned short* Fr = Fb + (size_t)(rowBase + m) * DIM;
    const unsigned short* Tr = Tb + (size_t)(colBase + m) * DIM;

    v8f c0 = {}, c1 = {}, c2 = {}, c3 = {};

    for (int k0 = 0; k0 < DIM; k0 += 32) {
        const int ka = k0 + kh * 8;
        const int kb = ka + 16;
        v16bf Bt = load_frag(Tr,                     ka, kb);
        v16bf A0 = load_frag(Fr,                     ka, kb);
        v16bf A1 = load_frag(Fr + (size_t)16 * DIM,  ka, kb);
        v16bf A2 = load_frag(Fr + (size_t)32 * DIM,  ka, kb);
        v16bf A3 = load_frag(Fr + (size_t)48 * DIM,  ka, kb);
        c0 = __builtin_amdgcn_wmma_f32_16x16x32_bf16(false, A0, false, Bt,
                                                     (short)0, c0, false, false);
        c1 = __builtin_amdgcn_wmma_f32_16x16x32_bf16(false, A1, false, Bt,
                                                     (short)0, c1, false, false);
        c2 = __builtin_amdgcn_wmma_f32_16x16x32_bf16(false, A2, false, Bt,
                                                     (short)0, c2, false, false);
        c3 = __builtin_amdgcn_wmma_f32_16x16x32_bf16(false, A3, false, Bt,
                                                     (short)0, c3, false, false);
    }

    // D layout: VGPR r holds (M = r + 8*(lane>>4), N = lane&15)
    const int nOut  = colBase + (lane & 15);
    const int mHalf = (lane >> 4) * 8;
    if (nOut < NC) {
        #pragma unroll
        for (int r = 0; r < 8; ++r) {
            const int rw = rowBase + mHalf + r;
            Km[(size_t)(rw     ) * NC + nOut] = gibbs(c0[r]);
            Km[(size_t)(rw + 16) * NC + nOut] = gibbs(c1[r]);
            Km[(size_t)(rw + 32) * NC + nOut] = gibbs(c2[r]);
            Km[(size_t)(rw + 48) * NC + nOut] = gibbs(c3[r]);
        }
    }
}

// ---------------------------------------------------------------------------
// init: u = 1/BS ; counter = 0
// ---------------------------------------------------------------------------
__global__ __launch_bounds__(256)
void init_kernel(float* __restrict__ u, unsigned int* __restrict__ counter) {
    const int i = blockIdx.x * 256 + threadIdx.x;
    if (i < BS) u[i] = 1.0f / (float)BS;
    if (i == 0) *counter = 0u;
}

// ---------------------------------------------------------------------------
// Column sums: partial[b, j] = sum over block b's 128 rows of Km[r, j]*u[r].
// Last block (counter pattern) reduces partials deterministically and emits
// v[j] = (ratios[j]*BS) / KtU[j], then resets counter. grid = 128 blocks.
// ---------------------------------------------------------------------------
__global__ __launch_bounds__(256)
void colsum_v_kernel(const float* __restrict__ Km, const float* __restrict__ u,
                     const float* __restrict__ ratios,
                     float* __restrict__ partial, float* __restrict__ v,
                     unsigned int* __restrict__ counter) {
    const int t  = threadIdx.x;
    const int b  = blockIdx.x;
    const int r0 = b * 128;
    float a0 = 0.f, a1 = 0.f, a2 = 0.f, a3 = 0.f;
    #pragma unroll 8
    for (int r = 0; r < 128; ++r) {
        const float  ur = u[r0 + r];
        const float* Kr = Km + (size_t)(r0 + r) * NC;
        if ((r & 7) == 0 && r + 8 < 128)                  // stream-ahead hint
            __builtin_prefetch(Kr + (size_t)8 * NC + t, 0, 1);
        a0 += Kr[t      ] * ur;
        a1 += Kr[t + 256] * ur;
        a2 += Kr[t + 512] * ur;
        if (t + 768 < NC) a3 += Kr[t + 768] * ur;
    }
    float* P = partial + (size_t)b * 1024;
    P[t] = a0; P[t + 256] = a1; P[t + 512] = a2;
    if (t + 768 < NC) P[t + 768] = a3;

    __shared__ int lastFlag;
    __threadfence();
    __syncthreads();
    if (t == 0) {
        unsigned int n = atomicAdd(counter, 1u);
        lastFlag = (n == 127u);
    }
    __syncthreads();
    if (lastFlag) {
        __threadfence();
        for (int j = t; j < NC; j += 256) {
            float s = 0.f;
            for (int bb = 0; bb < 128; ++bb) s += partial[(size_t)bb * 1024 + j];
            v[j] = (ratios[j] * (float)BS) / s;
        }
        if (t == 0) *counter = 0u;   // clean for next iteration / next call
    }
}

// ---------------------------------------------------------------------------
// Row sums: u[i] = 1 / sum_j Km[i,j]*v[j]. v staged in LDS; one wave handles
// 8 rows, lanes split columns (float4), in-wave shfl_xor reduction.
// grid = BS/64 = 256 blocks x 256 threads (8 waves).
// ---------------------------------------------------------------------------
__global__ __launch_bounds__(256)
void rowsum_u_kernel(const float* __restrict__ Km, const float* __restrict__ v,
                     float* __restrict__ u) {
    __shared__ __align__(16) float sv[NC];
    for (int j = threadIdx.x; j < NC; j += 256) sv[j] = v[j];
    __syncthreads();

    const int lane    = threadIdx.x & 31;
    const int wave    = threadIdx.x >> 5;
    const int rowBase = blockIdx.x * 64 + wave * 8;
    const float4* V4  = reinterpret_cast<const float4*>(sv);

    for (int rr = 0; rr < 8; ++rr) {
        const int row = rowBase + rr;
        const float4* K4 = reinterpret_cast<const float4*>(Km + (size_t)row * NC);
        if (rr + 1 < 8)                                   // stream-ahead hint
            __builtin_prefetch(Km + (size_t)(row + 1) * NC + lane * 4, 0, 1);
        float s = 0.f;
        for (int c4 = lane; c4 < NC / 4; c4 += 32) {
            float4 kv = K4[c4];
            float4 vv = V4[c4];
            s += kv.x * vv.x + kv.y * vv.y + kv.z * vv.z + kv.w * vv.w;
        }
        #pragma unroll
        for (int off = 16; off > 0; off >>= 1) s += __shfl_xor(s, off, 32);
        if (lane == 0) u[row] = 1.0f / s;
    }
}

// ---------------------------------------------------------------------------
// plan[i,j] = u[i] * Km[i,j] * v[j]   (float4; 1000 % 4 == 0, rows = 250 f4)
// grid = BS*NC/4/256 = 16000 blocks
// ---------------------------------------------------------------------------
__global__ __launch_bounds__(256)
void plan_kernel(const float* __restrict__ Km, const float* __restrict__ u,
                 const float* __restrict__ v, float* __restrict__ out) {
    const int f   = blockIdx.x * 256 + threadIdx.x;  // float4 index
    const int row = f / 250;
    const int c4  = f - row * 250;
    const float4 kv = reinterpret_cast<const float4*>(Km)[f];
    const float4 vv = reinterpret_cast<const float4*>(v)[c4];
    const float  ur = u[row];
    float4 o;
    o.x = ur * kv.x * vv.x; o.y = ur * kv.y * vv.y;
    o.z = ur * kv.z * vv.z; o.w = ur * kv.w * vv.w;
    reinterpret_cast<float4*>(out)[f] = o;
}

extern "C" void kernel_launch(void* const* d_in, const int* in_sizes, int n_in,
                              void* d_out, int out_size, void* d_ws, size_t ws_size,
                              hipStream_t stream) {
    (void)in_sizes; (void)n_in; (void)out_size; (void)ws_size;
    const float* features = (const float*)d_in[0];   // [BS, DIM]
    const float* text     = (const float*)d_in[1];   // [NC, DIM]
    const float* ratios   = (const float*)d_in[2];   // [NC]
    float* out = (float*)d_out;

    // workspace layout (floats): Km | u | v | partial | counter(+pad) | Fb | Tb
    float* Km      = (float*)d_ws;                       // BS*NC
    float* u       = Km + (size_t)BS * NC;               // BS
    float* v       = u + BS;                             // 1024 (NC used)
    float* partial = v + 1024;                           // 128 * 1024
    unsigned int* counter = (unsigned int*)(partial + 128 * 1024);
    unsigned int* Fb = counter + 16;                     // BS*DIM/2 packed bf16 pairs
    unsigned int* Tb = Fb + (size_t)BS * DIM / 2;        // NCP*DIM/2 packed bf16 pairs

    init_kernel<<<(BS + 255) / 256, 256, 0, stream>>>(u, counter);

    cvt_f_kernel<<<(BS * DIM / 2) / 256, 256, 0, stream>>>(features, Fb);
    cvt_t_kernel<<<(NCP * DIM / 2) / 256, 256, 0, stream>>>(text, Tb);

    dim3 gGemm(BS / 256, NCP / 16);                      // 64 x 63
    gemm_gibbs_kernel<<<gGemm, 128, 0, stream>>>(
        (const unsigned short*)Fb, (const unsigned short*)Tb, Km);

    for (int it = 0; it < NITER; ++it) {
        colsum_v_kernel<<<128, 256, 0, stream>>>(Km, u, ratios, partial, v, counter);
        rowsum_u_kernel<<<BS / 64, 256, 0, stream>>>(Km, v, u);
    }

    plan_kernel<<<(BS * NC / 4) / 256, 256, 0, stream>>>(Km, u, v, out);
}
